// WaveNetModel_18614388261637
// MI455X (gfx1250) — matrix operand order repaired and checked by reading, not compiled, not run on hardware
//
#include <hip/hip_runtime.h>
#include <hip/hip_bf16.h>

// ---------------------------------------------------------------------------
// WaveNet forward for MI455X (gfx1250, wave32, WMMA 16x16x32 bf16).
//
// All channel contractions are K=32 => exactly one V_WMMA_F32_16X16X32_BF16
// per 16x16 output tile. Activations live in [b][t][c] bf16 so a B operand
// (K=c, N=t) is two contiguous b128 loads per lane. Weights are pre-packed
// into the A-operand lane layout (one prep kernel per launch).
//
// Gated activation uses the gfx1250 hardware V_TANH_F32 (branchless, TRANS32,
// co-issues with WMMA) instead of libm's branchy tanhf.
//
// Skip accumulation only over the last 1024 steps (right-aligned in the
// reference); per-layer skip biases are folded into the accumulator init.
// ---------------------------------------------------------------------------

typedef __attribute__((ext_vector_type(16))) __bf16 v16bf;
typedef __attribute__((ext_vector_type(8)))  __bf16 v8bf;
typedef __attribute__((ext_vector_type(8)))  float  v8f;

#define NLAY 40
#define INPUT_LENGTH 5116
#define XROWS 5120            // padded time stride for activation buffers
#define OUTLEN 1024
#define NBATCH 16
#define LW_ELEMS 21504        // per-layer packed bf16: 5*1024 + 32*512
// packed layer block (bf16 element offsets):
//   f(tap0)=0  f(tap1)=1024  g(tap0)=2048  g(tap1)=3072  res=4096  skip=5120

__device__ inline v16bf load16bf(const __bf16* p) {
  union { float4 f[2]; v16bf v; } u;
  const float4* q = reinterpret_cast<const float4*>(p);
  u.f[0] = q[0]; u.f[1] = q[1];
  return u.v;
}
__device__ inline v8bf load8bf(const __bf16* p) {
  union { float4 f; v8bf v; } u;
  u.f = *reinterpret_cast<const float4*>(p);
  return u.v;
}
__device__ inline void store8bf(__bf16* p, v8bf v) {
  union { float4 f; v8bf v; } u; u.v = v;
  *reinterpret_cast<float4*>(p) = u.f;
}
__device__ inline v8f wmma_bf16(v16bf a, v16bf b, v8f c) {
  return __builtin_amdgcn_wmma_f32_16x16x32_bf16(false, a, false, b,
                                                 (short)0, c, false, false);
}

// Branchless tanh: gfx1250 has hardware V_TANH_F32 (TRANS32, co-issues with
// WMMA). Fallback is a single v_exp_f32 + rcp, still branchless.
__device__ inline float fast_tanh(float x) {
#if __has_builtin(__builtin_amdgcn_tanhf)
  return __builtin_amdgcn_tanhf(x);
#else
  float e = __expf(2.0f * x);
  return 1.0f - 2.0f / (e + 1.0f);
#endif
}
// sigmoid(x) = 0.5 + 0.5*tanh(x/2)
__device__ inline float fast_sigmoid(float x) {
  return 0.5f + 0.5f * fast_tanh(0.5f * x);
}

// A-operand column for lane/element (contiguous k = 16*(lane/16) + j, matching B)
__device__ inline int a_col(int lane, int j) {
  return j + 16 * (lane >> 4);
}

// --------------------------- weight packing --------------------------------
__global__ void prep_weights(const float* __restrict__ fw, const float* __restrict__ gw,
                             const float* __restrict__ rw, const float* __restrict__ sw,
                             const float* __restrict__ e0w, const float* __restrict__ e1w,
                             __bf16* __restrict__ packLW, __bf16* __restrict__ packE0,
                             __bf16* __restrict__ packE1) {
  int tid = blockIdx.x * blockDim.x + threadIdx.x;
  const int layerTotal = NLAY * LW_ELEMS;   // 860160
  const int e0Total = 16 * 16 * 512;        // 256x512, 16 mt x 16 kt tiles
  const int e1Total = 8 * 8 * 512;          // 128x256,  8 mt x  8 kt tiles
  if (tid < layerTotal) {
    int i = tid / LW_ELEMS, e = tid % LW_ELEMS;
    int seg, idx;
    if (e < 5120) { seg = e / 1024; idx = e % 1024; }
    else          { seg = 5;        idx = e - 5120; }
    int mt = idx / 512, r2 = idx % 512;
    int lane = r2 / 16, j = r2 % 16;
    int o = (lane & 15) + 16 * mt;
    int c = a_col(lane, j);
    float val;
    if      (seg == 0) val = fw[((i * 32 + o) * 32 + c) * 2 + 0];
    else if (seg == 1) val = fw[((i * 32 + o) * 32 + c) * 2 + 1];
    else if (seg == 2) val = gw[((i * 32 + o) * 32 + c) * 2 + 0];
    else if (seg == 3) val = gw[((i * 32 + o) * 32 + c) * 2 + 1];
    else if (seg == 4) val = rw[(i * 32 + o) * 32 + c];
    else               val = sw[(i * 512 + o) * 32 + c];
    packLW[tid] = (__bf16)val;
    return;
  }
  tid -= layerTotal;
  if (tid < e0Total) {
    int mt = tid / (16 * 512), r = tid % (16 * 512);
    int kt = r / 512, r2 = r % 512;
    int lane = r2 / 16, j = r2 % 16;
    int o = (lane & 15) + 16 * mt;
    int c = kt * 32 + a_col(lane, j);
    packE0[tid] = (__bf16)e0w[o * 512 + c];
    return;
  }
  tid -= e0Total;
  if (tid < e1Total) {
    int mt = tid / (8 * 512), r = tid % (8 * 512);
    int kt = r / 512, r2 = r % 512;
    int lane = r2 / 16, j = r2 % 16;
    int o = (lane & 15) + 16 * mt;
    int c = kt * 32 + a_col(lane, j);
    packE1[tid] = (__bf16)e1w[o * 256 + c];
  }
}

__global__ void sbsum_kernel(const float* __restrict__ skip_b, float* __restrict__ sbsum) {
  int o = blockIdx.x * blockDim.x + threadIdx.x;
  if (o < 512) {
    float s = 0.f;
    for (int i = 0; i < NLAY; ++i) s += skip_b[i * 512 + o];
    sbsum[o] = s;
  }
}

// start conv (32x1) + layout to [b][t][c] bf16
__global__ void start_kernel(const float* __restrict__ input, const float* __restrict__ sw,
                             const float* __restrict__ sb, __bf16* __restrict__ x0) {
  int tid = blockIdx.x * blockDim.x + threadIdx.x;
  if (tid >= NBATCH * INPUT_LENGTH) return;
  int b = tid / INPUT_LENGTH, t = tid % INPUT_LENGTH;
  float v = input[b * INPUT_LENGTH + t];
  __attribute__((aligned(16))) __bf16 row[32];
  for (int c = 0; c < 32; ++c) row[c] = (__bf16)(sw[c] * v + sb[c]);
  float4* dst = reinterpret_cast<float4*>(x0 + ((size_t)b * XROWS + t) * 32);
  const float4* src = reinterpret_cast<const float4*>(row);
  dst[0] = src[0]; dst[1] = src[1]; dst[2] = src[2]; dst[3] = src[3];
}

// skip accumulator := sum over layers of skip bias (bias folded out of layers)
__global__ void skip_init(const float* __restrict__ sbsum, float* __restrict__ skip_acc) {
  int tid = blockIdx.x * blockDim.x + threadIdx.x;
  if (tid >= NBATCH * OUTLEN * 128) return;
  int o4 = (tid & 127) * 4;
  size_t base = (size_t)(tid >> 7) * 512;
  *reinterpret_cast<float4*>(skip_acc + base + o4) =
      *reinterpret_cast<const float4*>(sbsum + o4);
}

// ------------------------------ layer kernel -------------------------------
// One wave handles one (batch b, 16-time tile). Tiles counted from the tail
// so the 64 skip-window tiles are exactly aligned.
__global__ void __launch_bounds__(128)
layer_kernel(const __bf16* __restrict__ xin, __bf16* __restrict__ xout,
             float* __restrict__ skip_acc, const __bf16* __restrict__ lw,
             const float* __restrict__ fb, const float* __restrict__ gb,
             const float* __restrict__ rb, int Lout, int d, int ntiles) {
  __shared__ __attribute__((aligned(64))) __bf16 hsh[4][16 * 32];
  int lane = threadIdx.x & 31;
  int wv = threadIdx.x >> 5;
  int wid = blockIdx.x * 4 + wv;
  if (wid >= NBATCH * ntiles) return;
  int b = wid / ntiles, k = wid % ntiles;
  int t0 = Lout - 16 * (k + 1);
  if (t0 < 0) t0 = 0;
  int tl = lane & 15, hf = lane >> 4;

  const __bf16* xb = xin + (size_t)b * XROWS * 32;
  // B operands: x[t0+tl][:] and x[t0+tl+d][:]  (taps 0 / 1 of the k=2 conv)
  v16bf B0 = load16bf(xb + (size_t)(t0 + tl) * 32 + 16 * hf);
  v16bf B1 = load16bf(xb + (size_t)(t0 + tl + d) * 32 + 16 * hf);

  // warm the skip-weight block (global_prefetch_b8) while filter/gate run
  if (k < 64) __builtin_prefetch(lw + 5120 + lane * 16, 0, 0);

  __bf16* hrow = &hsh[wv][tl * 32];
  v8f zero = {};
#pragma unroll
  for (int mt = 0; mt < 2; ++mt) {
    v16bf af0 = load16bf(lw + 0    + mt * 512 + lane * 16);
    v16bf af1 = load16bf(lw + 1024 + mt * 512 + lane * 16);
    v16bf ag0 = load16bf(lw + 2048 + mt * 512 + lane * 16);
    v16bf ag1 = load16bf(lw + 3072 + mt * 512 + lane * 16);
    v8f aF = wmma_bf16(af0, B0, zero); aF = wmma_bf16(af1, B1, aF);
    v8f aG = wmma_bf16(ag0, B0, zero); aG = wmma_bf16(ag1, B1, aG);
    const float* fbp = fb + mt * 16 + hf * 8;
    const float* gbp = gb + mt * 16 + hf * 8;
    v8bf hp;
#pragma unroll
    for (int r = 0; r < 8; ++r) {
      float fv = fast_tanh(aF[r] + fbp[r]);
      float gv = fast_sigmoid(aG[r] + gbp[r]);
      hp[r] = (__bf16)(fv * gv);
    }
    // D layout: channel = mt*16 + hf*8 + r, time = tl  -> restripe via LDS
    store8bf(hrow + mt * 16 + hf * 8, hp);
  }
  asm volatile("s_wait_dscnt 0" ::: "memory");   // intra-wave LDS RAW fence
  v16bf Bh = load16bf(&hsh[wv][tl * 32 + 16 * hf]);

  // residual path: xout = res_w @ h + res_b + x[t+d]
  __bf16* xo = xout + (size_t)b * XROWS * 32;
#pragma unroll
  for (int mt = 0; mt < 2; ++mt) {
    v16bf ar = load16bf(lw + 4096 + mt * 512 + lane * 16);
    v8f aR = wmma_bf16(ar, Bh, zero);
    const float* rbp = rb + mt * 16 + hf * 8;
    v8bf resid = load8bf(xb + (size_t)(t0 + tl + d) * 32 + mt * 16 + hf * 8);
    v8bf outp;
#pragma unroll
    for (int r = 0; r < 8; ++r)
      outp[r] = (__bf16)(aR[r] + rbp[r] + (float)resid[r]);
    store8bf(xo + (size_t)(t0 + tl) * 32 + mt * 16 + hf * 8, outp);
  }

  // skip path: only last 1024 steps survive to the head
  if (k < 64) {
    int tw = OUTLEN - 16 * (k + 1) + tl;
    float* sp = skip_acc + ((size_t)(b * OUTLEN + tw)) * 512 + hf * 8;
#pragma unroll 4
    for (int mt = 0; mt < 32; ++mt) {
      v16bf as = load16bf(lw + 5120 + mt * 512 + lane * 16);
      v8f aS = wmma_bf16(as, Bh, zero);
      float* q = sp + mt * 16;
      union { float4 f[2]; float s[8]; } u;
      u.f[0] = *reinterpret_cast<float4*>(q);
      u.f[1] = *reinterpret_cast<float4*>(q + 4);
#pragma unroll
      for (int r = 0; r < 8; ++r) u.s[r] += aS[r];
      *reinterpret_cast<float4*>(q)     = u.f[0];
      *reinterpret_cast<float4*>(q + 4) = u.f[1];
    }
  }
}

// ---------------------------- head GEMM kernel -----------------------------
// out[b][t][Cout] = bias + A @ relu(in[b][t][Cin]); wave = (b, 16-t tile,
// group of 4 M-tiles). relu folded into the bf16 B conversion.
__global__ void __launch_bounds__(128)
end_gemm(const float* __restrict__ in, float* __restrict__ out,
         const __bf16* __restrict__ packA, const float* __restrict__ bias,
         int Cin, int Cout, int nkt, int ngrp) {
  int lane = threadIdx.x & 31;
  int wv = threadIdx.x >> 5;
  int wid = blockIdx.x * 4 + wv;
  if (wid >= NBATCH * 64 * ngrp) return;
  int g = wid % ngrp;
  int k = (wid / ngrp) % 64;
  int b = wid / (ngrp * 64);
  int tl = lane & 15, hf = lane >> 4;
  int t = 16 * k + tl;
  const float* brow = in + ((size_t)b * OUTLEN + t) * Cin + 16 * hf;
  v8f acc[4] = {};
  for (int kt = 0; kt < nkt; ++kt) {
    const float* p = brow + kt * 32;
    if (kt + 1 < nkt) __builtin_prefetch(brow + (kt + 1) * 32, 0, 0);
    union { float4 f[4]; float s[16]; } u;
    const float4* pq = reinterpret_cast<const float4*>(p);
    u.f[0] = pq[0]; u.f[1] = pq[1]; u.f[2] = pq[2]; u.f[3] = pq[3];
    v16bf Bt;
#pragma unroll
    for (int e = 0; e < 16; ++e) Bt[e] = (__bf16)fmaxf(u.s[e], 0.0f);
#pragma unroll
    for (int m = 0; m < 4; ++m) {
      int mt = g * 4 + m;
      v16bf A = load16bf(packA + ((size_t)(mt * nkt + kt)) * 512 + lane * 16);
      acc[m] = wmma_bf16(A, Bt, acc[m]);
    }
  }
#pragma unroll
  for (int m = 0; m < 4; ++m) {
    int mt = g * 4 + m;
    const float* bp = bias + mt * 16 + hf * 8;
    float* orow = out + ((size_t)b * OUTLEN + t) * Cout + mt * 16 + hf * 8;
    union { float4 f[2]; float s[8]; } u;
#pragma unroll
    for (int r = 0; r < 8; ++r) u.s[r] = acc[m][r] + bp[r];
    reinterpret_cast<float4*>(orow)[0] = u.f[0];
    reinterpret_cast<float4*>(orow)[1] = u.f[1];
  }
}

// final 128 -> 1 projection
__global__ void final_kernel(const float* __restrict__ e1, const float* __restrict__ w2,
                             const float* __restrict__ b2, float* __restrict__ outp) {
  int tid = blockIdx.x * blockDim.x + threadIdx.x;
  if (tid >= NBATCH * OUTLEN) return;
  const float* row = e1 + (size_t)tid * 128;
  float acc = b2[0];
  for (int c = 0; c < 128; ++c) acc += w2[c] * fmaxf(row[c], 0.0f);
  outp[tid] = acc;
}

// ------------------------------- launcher ----------------------------------
extern "C" void kernel_launch(void* const* d_in, const int* in_sizes, int n_in,
                              void* d_out, int out_size, void* d_ws, size_t ws_size,
                              hipStream_t stream) {
  (void)in_sizes; (void)n_in; (void)out_size; (void)ws_size;
  const float* input    = (const float*)d_in[0];
  const float* start_w  = (const float*)d_in[1];
  const float* start_b  = (const float*)d_in[2];
  const float* filter_w = (const float*)d_in[3];
  const float* filter_b = (const float*)d_in[4];
  const float* gate_w   = (const float*)d_in[5];
  const float* gate_b   = (const float*)d_in[6];
  const float* res_w    = (const float*)d_in[7];
  const float* res_b    = (const float*)d_in[8];
  const float* skip_w   = (const float*)d_in[9];
  const float* skip_b   = (const float*)d_in[10];
  const float* end_w0   = (const float*)d_in[11];
  const float* end_b0   = (const float*)d_in[12];
  const float* end_w1   = (const float*)d_in[13];
  const float* end_b1   = (const float*)d_in[14];
  const float* end_w2   = (const float*)d_in[15];
  const float* end_b2   = (const float*)d_in[16];

  // workspace carve (~63 MB total)
  char* ws = (char*)d_ws;
  size_t off = 0;
  auto carve = [&](size_t bytes) -> char* {
    char* p = ws + off;
    off += (bytes + 255) & ~(size_t)255;
    return p;
  };
  float*  sbsum  = (float*) carve(512 * 4);
  __bf16* packLW = (__bf16*)carve((size_t)NLAY * LW_ELEMS * 2);
  __bf16* packE0 = (__bf16*)carve((size_t)16 * 16 * 512 * 2);
  __bf16* packE1 = (__bf16*)carve((size_t)8 * 8 * 512 * 2);
  __bf16* xA     = (__bf16*)carve((size_t)NBATCH * XROWS * 32 * 2);
  __bf16* xB     = (__bf16*)carve((size_t)NBATCH * XROWS * 32 * 2);
  float*  skip   = (float*) carve((size_t)NBATCH * OUTLEN * 512 * 4);
  float*  e0     = (float*) carve((size_t)NBATCH * OUTLEN * 256 * 4);
  float*  e1     = skip;  // alias: skip is dead after the first head GEMM

  {
    int total = NLAY * LW_ELEMS + 16 * 16 * 512 + 8 * 8 * 512;
    prep_weights<<<(total + 255) / 256, 256, 0, stream>>>(
        filter_w, gate_w, res_w, skip_w, end_w0, end_w1, packLW, packE0, packE1);
  }
  sbsum_kernel<<<2, 256, 0, stream>>>(skip_b, sbsum);
  start_kernel<<<(NBATCH * INPUT_LENGTH + 255) / 256, 256, 0, stream>>>(
      input, start_w, start_b, xA);
  skip_init<<<(NBATCH * OUTLEN * 128 + 255) / 256, 256, 0, stream>>>(sbsum, skip);

  __bf16* xin = xA;
  __bf16* xout = xB;
  int L = INPUT_LENGTH;
  int li = 0;
  for (int blk = 0; blk < 4; ++blk) {
    for (int j = 0; j < 10; ++j) {
      int d = 1 << j;
      int Lout = L - d;
      int ntiles = (Lout + 15) / 16;
      int waves = NBATCH * ntiles;
      layer_kernel<<<(waves + 3) / 4, 128, 0, stream>>>(
          xin, xout, skip, packLW + (size_t)li * LW_ELEMS,
          filter_b + li * 32, gate_b + li * 32, res_b + li * 32,
          Lout, d, ntiles);
      __bf16* t = xin; xin = xout; xout = t;
      L = Lout; ++li;
    }
  }
  // head: 512 -> 256 -> 128 -> 1 over the 1024-step window
  end_gemm<<<(NBATCH * 64 * 4 + 3) / 4 /*=1024*/, 128, 0, stream>>>(
      skip, e0, packE0, end_b0, 512, 256, 16, 4);
  end_gemm<<<(NBATCH * 64 * 2 + 3) / 4 /*=512*/, 128, 0, stream>>>(
      e0, e1, packE1, end_b1, 256, 128, 8, 2);
  final_kernel<<<(NBATCH * OUTLEN + 255) / 256, 256, 0, stream>>>(
      e1, end_w2, end_b2, (float*)d_out);
}
